// DMOR_80461917323301
// MI455X (gfx1250) — matrix operand order repaired and checked by reading, not compiled, not run on hardware
//
#include <hip/hip_runtime.h>
#include <hip/hip_bf16.h>
#include <math.h>

typedef __attribute__((ext_vector_type(16))) _Float16 v16h;
typedef __attribute__((ext_vector_type(8)))  _Float16 v8h;
typedef __attribute__((ext_vector_type(8)))  float    v8f;

#define BB   8
#define CC   64
#define HH   128
#define WW   128
#define HID  16
#define NOPS 5
#define KK   576            // C * 9, reordered as k = (kh*3+kw)*64 + c
#define TW   64             // positions per workgroup (one row segment)
#define TCOLS 68            // TW + 2*2 halo
#define TROWS 5             // h-2 .. h+2
#define CPAD 72             // channel stride in LDS tile (bank-conflict-free, 16B aligned)

// ---------------------------------------------------------------------------
// Kernel 1: repack conv weights to f16 with k = (kh*3+kw)*64 + c ordering.
// wf16 layout: [conv(2)][o(64)][k(576)]
// ---------------------------------------------------------------------------
__global__ __launch_bounds__(256) void dmor_prep_weights(
    const float* __restrict__ w3, const float* __restrict__ wd,
    _Float16* __restrict__ wf16)
{
    int idx = blockIdx.x * 256 + threadIdx.x;            // < 2*64*576 = 73728
    if (idx >= 2 * CC * KK) return;
    int conv = idx / (CC * KK);
    int rem  = idx - conv * (CC * KK);
    int o    = rem / KK;
    int k    = rem - o * KK;
    int r    = k >> 6;                                   // kh*3+kw, 0..8
    int c    = k & 63;
    int kh   = r / 3;
    int kw   = r - kh * 3;
    const float* src = conv ? wd : w3;
    wf16[idx] = (_Float16)src[(o * CC + c) * 9 + kh * 3 + kw];
}

// ---------------------------------------------------------------------------
// Kernel 2: global average pool. One block per (b,c) plane.
// ---------------------------------------------------------------------------
__global__ __launch_bounds__(256) void dmor_gap(
    const float* __restrict__ x, float* __restrict__ xg)
{
    int bc = blockIdx.x;                                 // 0..511
    const float* p = x + (size_t)bc * (HH * WW);
    float s = 0.f;
    for (int i = threadIdx.x; i < HH * WW; i += 256) s += p[i];
    __shared__ float red[256];
    red[threadIdx.x] = s;
    __syncthreads();
    for (int st = 128; st > 0; st >>= 1) {
        if (threadIdx.x < st) red[threadIdx.x] += red[threadIdx.x + st];
        __syncthreads();
    }
    if (threadIdx.x == 0) xg[bc] = red[0] * (1.f / (HH * WW));
}

// ---------------------------------------------------------------------------
// Kernel 3: global router MLP (single block).
// ---------------------------------------------------------------------------
__global__ __launch_bounds__(256) void dmor_router(
    const float* __restrict__ xg, const float* __restrict__ gr_w1,
    const float* __restrict__ gr_w2, const float* __restrict__ gr_b2,
    float* __restrict__ wgl)
{
    __shared__ float hbuf[BB * HID];
    int t = threadIdx.x;
    if (t < BB * HID) {
        int b = t >> 4, j = t & 15;
        float a = 0.f;
        for (int c = 0; c < CC; ++c) a += gr_w1[j * CC + c] * xg[b * CC + c];
        hbuf[t] = fmaxf(a, 0.f);
    }
    __syncthreads();
    if (t < BB * NOPS) {
        int b = t / NOPS, n = t - b * NOPS;
        float a = gr_b2[n];
        for (int j = 0; j < HID; ++j) a += gr_w2[n * HID + j] * hbuf[b * HID + j];
        wgl[t] = a;
    }
}

// ---------------------------------------------------------------------------
// Kernel 4: fused DMOR. One workgroup (8 waves) per 64-wide row segment.
//   LDS x tile layout: [row(5)][col(68)][c(72 padded)] as v8h blocks so each
//   A-fragment is two contiguous ds_load_b128 per lane.
// ---------------------------------------------------------------------------
__global__ __launch_bounds__(256) void dmor_main(
    const float*    __restrict__ x,
    const _Float16* __restrict__ wf16,
    const float* __restrict__ b3,  const float* __restrict__ bd,
    const float* __restrict__ wdw, const float* __restrict__ bdw,
    const float* __restrict__ sr_w, const float* __restrict__ sr_b,
    const float* __restrict__ wgl,
    float* __restrict__ out)
{
    const int w0  = blockIdx.x * TW;     // 0 or 64
    const int h   = blockIdx.y;          // 0..127
    const int b   = blockIdx.z;          // 0..7
    const int tid = threadIdx.x;

    __shared__ v8h      xt8[TROWS * TCOLS * (CPAD / 8)];   // 48960 B
    __shared__ _Float16 convbuf[2 * TW * CC];              // 16384 B
    _Float16* xt = (_Float16*)xt8;

    const float* xb = x + (size_t)b * CC * HH * WW;

    // ---- stage f16 halo tile of x: xt[(row*68+col)*72 + c] ----
    for (int i = tid; i < CC * TROWS * TCOLS; i += 256) {
        int c   = i / (TROWS * TCOLS);
        int rem = i - c * (TROWS * TCOLS);
        int row = rem / TCOLS;
        int col = rem - row * TCOLS;
        int gh  = h + row - 2;
        int gw  = w0 + col - 2;
        float v = 0.f;
        if ((unsigned)gh < (unsigned)HH && (unsigned)gw < (unsigned)WW)
            v = xb[c * (HH * WW) + gh * WW + gw];
        xt[(row * TCOLS + col) * CPAD + c] = (_Float16)v;
    }
    __syncthreads();

    // ---- WMMA phase ----
    const int wave  = tid >> 5;          // 0..7
    const int lane  = tid & 31;
    const int conv  = wave >> 2;         // 0: conv3x3, 1: dilated (d=2)
    const int ntile = wave & 3;
    const int d     = conv ? 2 : 1;
    const int m     = lane & 15;         // A-row within tile / B column
    const int khalf = lane >> 4;         // 0,1
    const int ch    = ntile * 16 + m;    // output channel this lane owns in B/D

    const _Float16* wrow = wf16 + (size_t)conv * (CC * KK)
                                + (size_t)ch * KK + khalf * 16;

    v8f acc[4] = {};                     // 4 M-tiles of 16 positions each

    #pragma unroll 1
    for (int chunk = 0; chunk < 18; ++chunk) {
        // B fragment: 16 contiguous halfs per lane (32B aligned)
        v16h bfrag = *(const v16h*)(wrow + chunk * 32);
        __builtin_prefetch(wrow + (chunk + 1) * 32, 0, 0);

        // Two runs of 8 consecutive K per lane -> two v8h LDS block indices
        int a8[2];
        #pragma unroll
        for (int rr = 0; rr < 2; ++rr) {
            int K   = chunk * 32 + khalf * 16 + rr * 8;
            int r   = K >> 6;            // kh*3+kw (run never crosses r)
            int c0  = K & 63;            // multiple of 8
            int kh  = r / 3;
            int kw  = r - kh * 3;
            int row = 2 + d * (kh - 1);
            int col = 2 + m + d * (kw - 1);
            a8[rr]  = ((row * TCOLS + col) * CPAD + c0) >> 3;   // v8h units
        }

        #pragma unroll
        for (int mt = 0; mt < 4; ++mt) {
            // +16 positions per M-tile => +16*CPAD halfs = +144 v8h blocks
            v8h lo = xt8[a8[0] + mt * (16 * CPAD / 8)];
            v8h hi = xt8[a8[1] + mt * (16 * CPAD / 8)];
            v16h afrag = __builtin_shufflevector(
                lo, hi, 0, 1, 2, 3, 4, 5, 6, 7, 8, 9, 10, 11, 12, 13, 14, 15);
            acc[mt] = __builtin_amdgcn_wmma_f32_16x16x32_f16(
                false, afrag, false, bfrag, (short)0, acc[mt], false, false);
        }
    }

    // bias + stash conv results in LDS as f16: convbuf[conv][pos][ch]
    {
        const float bias = (conv ? bd : b3)[ch];
        #pragma unroll
        for (int mt = 0; mt < 4; ++mt) {
            #pragma unroll
            for (int r = 0; r < 8; ++r) {
                int pos = mt * 16 + r + khalf * 8;   // C/D layout: M = r + 8*(lane/16)
                convbuf[(conv * TW + pos) * CC + ch] = (_Float16)(acc[mt][r] + bias);
            }
        }
    }
    __syncthreads();

    // ---- routing + blend phase ----
    const int p  = tid & 63;             // position in segment
    const int cg = tid >> 6;             // channel group 0..3
    const int gw = w0 + p;

    // spatial + global router logits (fp32 from global, L2-hot)
    float logit[NOPS];
    #pragma unroll
    for (int n = 0; n < NOPS; ++n) logit[n] = wgl[b * NOPS + n] + sr_b[n];
    for (int c = 0; c < CC; ++c) {
        float xv = xb[c * (HH * WW) + h * WW + gw];
        #pragma unroll
        for (int n = 0; n < NOPS; ++n) logit[n] += sr_w[n * CC + c] * xv;
    }

    // softmax over 5
    float mx = logit[0];
    #pragma unroll
    for (int n = 1; n < NOPS; ++n) mx = fmaxf(mx, logit[n]);
    float wsm[NOPS]; float ssum = 0.f;
    #pragma unroll
    for (int n = 0; n < NOPS; ++n) { wsm[n] = __expf(logit[n] - mx); ssum += wsm[n]; }
    float rs = 1.f / ssum;
    #pragma unroll
    for (int n = 0; n < NOPS; ++n) wsm[n] *= rs;

    // top-2 (first occurrence wins ties, matching lax.top_k), renorm with EPS
    float v0 = wsm[0]; int i0 = 0;
    #pragma unroll
    for (int n = 1; n < NOPS; ++n) { bool g = wsm[n] > v0; v0 = g ? wsm[n] : v0; i0 = g ? n : i0; }
    float v1 = -1.f; int i1 = -1;
    #pragma unroll
    for (int n = 0; n < NOPS; ++n) {
        bool g = (n != i0) && (wsm[n] > v1);
        v1 = g ? wsm[n] : v1; i1 = g ? n : i1;
    }
    float inv = 1.f / (v0 + v1 + 1e-6f);
    float wt[NOPS];
    #pragma unroll
    for (int n = 0; n < NOPS; ++n)
        wt[n] = (n == i0 || n == i1) ? wsm[n] * inv : 0.f;

    // blend 16 channels per thread
    #pragma unroll 1
    for (int i = 0; i < 16; ++i) {
        int c = cg * 16 + i;
        const float* xc = xb + c * (HH * WW);
        float ident = xc[h * WW + gw];
        float sacc = 0.f, dwacc = bdw[c];
        #pragma unroll
        for (int kh = 0; kh < 3; ++kh) {
            int gh2 = h + kh - 1;
            #pragma unroll
            for (int kw = 0; kw < 3; ++kw) {
                int gw2 = gw + kw - 1;
                float v = 0.f;
                if ((unsigned)gh2 < (unsigned)HH && (unsigned)gw2 < (unsigned)WW)
                    v = xc[gh2 * WW + gw2];
                sacc  += v;
                dwacc += v * wdw[c * 9 + kh * 3 + kw];
            }
        }
        float avg = sacc * (1.f / 9.f);
        float c3  = (float)convbuf[(0 * TW + p) * CC + c];
        float cd  = (float)convbuf[(1 * TW + p) * CC + c];
        float o = wt[0] * ident + wt[1] * c3 + wt[2] * cd + wt[3] * avg + wt[4] * dwacc;
        out[((size_t)b * CC + c) * (HH * WW) + h * WW + gw] = o;
    }
}

// ---------------------------------------------------------------------------
extern "C" void kernel_launch(void* const* d_in, const int* in_sizes, int n_in,
                              void* d_out, int out_size, void* d_ws, size_t ws_size,
                              hipStream_t stream) {
    const float* x     = (const float*)d_in[0];
    const float* w3    = (const float*)d_in[1];
    const float* b3    = (const float*)d_in[2];
    const float* wd    = (const float*)d_in[3];
    const float* bd    = (const float*)d_in[4];
    const float* wdw   = (const float*)d_in[5];
    const float* bdw   = (const float*)d_in[6];
    const float* gr_w1 = (const float*)d_in[7];
    const float* gr_w2 = (const float*)d_in[8];
    const float* gr_b2 = (const float*)d_in[9];
    const float* sr_w  = (const float*)d_in[10];
    const float* sr_b  = (const float*)d_in[11];
    float* out = (float*)d_out;

    // workspace layout
    _Float16* wf16 = (_Float16*)d_ws;                         // 2*64*576 halfs = 147456 B
    float*    xg   = (float*)((char*)d_ws + 2 * CC * KK * 2); // 512 floats
    float*    wgl  = xg + BB * CC;                            // 40 floats

    dmor_prep_weights<<<(2 * CC * KK + 255) / 256, 256, 0, stream>>>(w3, wd, wf16);
    dmor_gap<<<BB * CC, 256, 0, stream>>>(x, xg);
    dmor_router<<<1, 256, 0, stream>>>(xg, gr_w1, gr_w2, gr_b2, wgl);

    dim3 grid(WW / TW, HH, BB);   // (2, 128, 8)
    dmor_main<<<grid, 256, 0, stream>>>(x, wf16, b3, bd, wdw, bdw,
                                        sr_w, sr_b, wgl, out);
}
